// ReformerEnc_4698694222592
// MI455X (gfx1250) — matrix operand order repaired and checked by reading, not compiled, not run on hardware
//
#include <hip/hip_runtime.h>

// ---------------------------------------------------------------------------
// Reformer encoder (2 layers, reversible) for MI455X / gfx1250.
// B=2, T=4096, DIM=1024, HEADS=8, DH=128, BUCKET=64, NHASH=4, NBUCKETS=64.
// GEMMs + attention matmuls: v_wmma_f32_16x16x32_bf16 (f32 accum).
// GEMM tile staging: global_load_async_to_lds_b128 (A) + TDM tensor_load (B).
// ---------------------------------------------------------------------------

#define BB     2
#define TT     4096
#define DIMD   1024
#define HEADS  8
#define DH     128
#define NHASH  4
#define NBUCK  64
#define BUCKET 64
#define NCHUNK (NHASH * NBUCK)          // 256 chunks per bh row
#define BH     (BB * HEADS)             // 16
#define MTOK   (BB * TT)                // 8192
#define SROW   (NHASH * TT)             // 16384 sorted elements per bh
#define SCALE_QK 0.08838834764831845f   // 128^-0.5
#define SELF_VAL (-5e4f)

typedef __attribute__((ext_vector_type(16))) __bf16 bf16x16;
typedef __attribute__((ext_vector_type(8)))  float  f32x8;
typedef unsigned int u32x4 __attribute__((ext_vector_type(4)));
typedef int          i32x4 __attribute__((ext_vector_type(4)));
typedef int          i32x8 __attribute__((ext_vector_type(8)));

struct U128 { unsigned int a, b, c, d; };
union FragU { U128 u[2]; bf16x16 v; };

__device__ inline unsigned short f32_to_bf16(float f) {
    union { float f; unsigned int u; } c; c.f = f;
    unsigned int u = c.u;
    return (unsigned short)((u + 0x7FFFu + ((u >> 16) & 1u)) >> 16);
}

__device__ inline float gelu_exact(float x) {
    return 0.5f * x * (1.0f + erff(x * 0.70710678118654752440f));
}

// Async direct-to-LDS 16B copy (per-lane), tracked by ASYNCcnt.
__device__ inline void async_copy_b128(unsigned lds_addr, const void* gptr) {
    asm volatile("global_load_async_to_lds_b128 %0, %1, off"
                 :: "v"(lds_addr), "v"((unsigned long long)(size_t)gptr)
                 : "memory");
}
__device__ inline void wait_async0() {
    asm volatile("s_wait_asynccnt 0x0" ::: "memory");
}

// A fragment: 16x32 bf16, row-major LDS/gmem tile with 'stride' elems per row.
// lane<16: K {k0..k0+7, k0+16..k0+23}; lane>=16: +8 on each chunk.
__device__ inline bf16x16 load_frag_a(const unsigned short* base, int stride,
                                      int rowBase, int k0, int lane) {
    int l16 = lane & 15, hi = lane >> 4;
    const unsigned short* p = base + (rowBase + l16) * stride + k0 + hi * 8;
    FragU f;
    f.u[0] = *(const U128*)(p);
    f.u[1] = *(const U128*)(p + 16);
    return f.v;
}

// B fragment: 32x16 bf16 (KxN), stored as Bt[n][k]: lane<16: K {k0..k0+7, k0+16..k0+23}; lane>=16: +8.
__device__ inline bf16x16 load_frag_b(const unsigned short* base, int stride,
                                      int colBase, int k0, int lane) {
    int l16 = lane & 15, hi = lane >> 4;
    const unsigned short* p = base + (colBase + l16) * stride + k0 + hi * 8;
    FragU f;
    f.u[0] = *(const U128*)(p);
    f.u[1] = *(const U128*)(p + 16);
    return f.v;
}

// ---------------------------------------------------------------------------
// Weight convert + transpose: in f32 [K][N] -> out bf16 [N][K]
// ---------------------------------------------------------------------------
__global__ __launch_bounds__(256) void convt_kernel(const float* __restrict__ in,
                                                    unsigned short* __restrict__ out,
                                                    int K, int N) {
    long long idx = (long long)blockIdx.x * 256 + threadIdx.x;
    if (idx >= (long long)K * N) return;
    int n = (int)(idx / K), k = (int)(idx % K);
    out[idx] = f32_to_bf16(in[(long long)k * N + n]);
}

// ---------------------------------------------------------------------------
// LayerNorm: one block per token row, 256 thr x 4 elems, output bf16
// ---------------------------------------------------------------------------
__global__ __launch_bounds__(256) void ln_kernel(const float* __restrict__ x,
                                                 const float* __restrict__ g,
                                                 const float* __restrict__ bt,
                                                 unsigned short* __restrict__ out) {
    int row = blockIdx.x, tid = threadIdx.x;
    const float* xr = x + (size_t)row * DIMD;
    float4 v = *(const float4*)(xr + tid * 4);
    float s1 = v.x + v.y + v.z + v.w;
    float s2 = v.x * v.x + v.y * v.y + v.z * v.z + v.w * v.w;
    __shared__ float r1[256], r2[256];
    r1[tid] = s1; r2[tid] = s2;
    __syncthreads();
    for (int o = 128; o > 0; o >>= 1) {
        if (tid < o) { r1[tid] += r1[tid + o]; r2[tid] += r2[tid + o]; }
        __syncthreads();
    }
    float mean = r1[0] * (1.0f / DIMD);
    float var  = r2[0] * (1.0f / DIMD) - mean * mean;
    float rstd = rsqrtf(var + 1e-5f);
    int base = tid * 4;
    float e[4] = { v.x, v.y, v.z, v.w };
    unsigned short* orow = out + (size_t)row * DIMD + base;
    for (int i = 0; i < 4; ++i)
        orow[i] = f32_to_bf16((e[i] - mean) * rstd * g[base + i] + bt[base + i]);
}

// ---------------------------------------------------------------------------
// WMMA bf16 GEMM: C[M,N](f32/bf16) = A[M,K]bf16 * Bt[N,K]bf16 (+bias,+act,+res)
// Block tile 128x128, 8 waves, each wave 32x64 (2x4 accum tiles of 16x16).
// A tile -> LDS via global_load_async_to_lds_b128 (ASYNCcnt),
// B tile -> LDS via TDM tensor_load_to_lds (TENSORcnt), issued by wave 0.
// ---------------------------------------------------------------------------
__global__ __launch_bounds__(256) void gemm_bf16_kernel(
    const unsigned short* __restrict__ A, const unsigned short* __restrict__ Bt,
    const float* __restrict__ bias, const float* __restrict__ residual,
    float* __restrict__ Cf, unsigned short* __restrict__ Cbf,
    int M, int N, int K, int act) {

    __shared__ __align__(16) unsigned short As[128 * 32];
    __shared__ __align__(16) unsigned short Bs[128 * 32];

    int tid = threadIdx.x, lane = tid & 31, w = tid >> 5;
    int m0 = blockIdx.y * 128, n0 = blockIdx.x * 128;
    int wm = (w & 3) * 32, wn = (w >> 2) * 64;

    unsigned asBase = (unsigned)(size_t)(void*)As;   // LDS byte offsets
    unsigned bsBase = (unsigned)(size_t)(void*)Bs;

    f32x8 acc[2][4];
#pragma unroll
    for (int i = 0; i < 2; ++i)
#pragma unroll
        for (int j = 0; j < 4; ++j)
#pragma unroll
            for (int e = 0; e < 8; ++e) acc[i][j][e] = 0.0f;

    for (int k0 = 0; k0 < K; k0 += 32) {
        // ---- A tile: 128x32 bf16, 512 async b128 transfers (2 per lane) ----
#pragma unroll
        for (int s = 0; s < 2; ++s) {
            int li = tid + s * 256;
            int r = li >> 2, q = li & 3;
            async_copy_b128(asBase + (unsigned)((r * 32 + q * 8) * 2),
                            A + (size_t)(m0 + r) * K + k0 + q * 8);
        }
        // ---- B tile: 128x32 bf16 via one TDM descriptor (wave 0) -----------
        if (w == 0) {
            unsigned long long gaB =
                (unsigned long long)(size_t)(Bt + (size_t)n0 * K + k0);
            u32x4 g0;
            g0[0] = 1u;                                    // count=1, user D#
            g0[1] = bsBase;                                // lds_addr
            g0[2] = (unsigned)(gaB & 0xFFFFFFFFu);         // global_addr[31:0]
            g0[3] = (unsigned)((gaB >> 32) & 0x01FFFFFFu)  // global_addr[56:32]
                  | (2u << 30);                            // type = 2 (image)
            i32x8 g1;
            g1[0] = (int)(1u << 16);                       // data_size = 2B
            g1[1] = (int)(((unsigned)K & 0xFFFFu) << 16);  // tensor_dim0 lo16
            g1[2] = (int)((((unsigned)K >> 16) & 0xFFFFu)
                  | (((unsigned)N & 0xFFFFu) << 16));      // dim0 hi / dim1 lo
            g1[3] = (int)((((unsigned)N >> 16) & 0xFFFFu)
                  | (32u << 16));                          // dim1 hi / tile_dim0=32
            g1[4] = 128;                                   // tile_dim1=128, tile_dim2=0
            g1[5] = (int)(unsigned)K;                      // tensor_dim0_stride
            g1[6] = 0;
            g1[7] = 0;
            i32x4 z4;
            z4[0] = 0; z4[1] = 0; z4[2] = 0; z4[3] = 0;
            i32x8 z8;
            z8[0] = 0; z8[1] = 0; z8[2] = 0; z8[3] = 0;
            z8[4] = 0; z8[5] = 0; z8[6] = 0; z8[7] = 0;
            __builtin_amdgcn_tensor_load_to_lds(g0, g1, z4, z4, z8, 0);
            __builtin_amdgcn_s_wait_tensorcnt(0);
        }
        if (k0 + 32 < K)                                   // global_prefetch_b8
            __builtin_prefetch(A + (size_t)(m0 + (tid >> 1)) * K + k0 + 32, 0, 1);
        wait_async0();
        __syncthreads();

        bf16x16 afr[2], bfr[4];
#pragma unroll
        for (int mt = 0; mt < 2; ++mt) afr[mt] = load_frag_a(As, 32, wm + mt * 16, 0, lane);
#pragma unroll
        for (int nt = 0; nt < 4; ++nt) bfr[nt] = load_frag_b(Bs, 32, wn + nt * 16, 0, lane);
#pragma unroll
        for (int mt = 0; mt < 2; ++mt)
#pragma unroll
            for (int nt = 0; nt < 4; ++nt)
                acc[mt][nt] = __builtin_amdgcn_wmma_f32_16x16x32_bf16(
                    false, afr[mt], false, bfr[nt], (short)0, acc[mt][nt], false, false);
        __syncthreads();
    }

    int l16 = lane & 15, hi = lane >> 4;
#pragma unroll
    for (int mt = 0; mt < 2; ++mt)
#pragma unroll
        for (int nt = 0; nt < 4; ++nt) {
            int rbase = m0 + wm + mt * 16 + hi * 8;
            int col   = n0 + wn + nt * 16 + l16;
            float bv = bias ? bias[col] : 0.0f;
#pragma unroll
            for (int e = 0; e < 8; ++e) {
                int row = rbase + e;
                float v = acc[mt][nt][e] + bv;
                if (act == 1) v = gelu_exact(v);
                if (residual) v += residual[(size_t)row * N + col];
                if (Cf)  Cf [(size_t)row * N + col] = v;
                if (Cbf) Cbf[(size_t)row * N + col] = f32_to_bf16(v);
            }
        }
}

// ---------------------------------------------------------------------------
// LSH hashing: block per (bh, t), 128 threads. rot layout [DH][NHASH][32].
// ---------------------------------------------------------------------------
__global__ __launch_bounds__(128) void hash_kernel(const float* __restrict__ qkproj,
                                                   const float* __restrict__ rot,
                                                   int* __restrict__ buckets) {
    int gid = blockIdx.x;
    int bhead = gid >> 12, t = gid & (TT - 1);
    int b = bhead >> 3, head = bhead & 7;
    int tid = threadIdx.x;
    __shared__ float q[DH];
    __shared__ float rv[128];
    q[tid] = qkproj[(size_t)(b * TT + t) * DIMD + head * DH + tid];
    __syncthreads();
    int h = tid >> 5, i = tid & 31;
    float r = 0.0f;
    for (int f = 0; f < DH; ++f) r += q[f] * rot[(f * NHASH + h) * 32 + i];
    rv[tid] = r;
    __syncthreads();
    if (i == 0) {
        float best = rv[h * 32]; int bi = 0;
        for (int k = 1; k < 32; ++k) { float v = rv[h * 32 + k]; if (v > best) { best = v; bi = k; } }
        for (int k = 0; k < 32; ++k) { float v = -rv[h * 32 + k]; if (v > best) { best = v; bi = k + 32; } }
        buckets[(size_t)bhead * SROW + h * TT + t] = h * NBUCK + bi;
    }
}

// ---------------------------------------------------------------------------
// Counting sort: histogram, exclusive scan, stable rank+scatter
// ---------------------------------------------------------------------------
__global__ __launch_bounds__(256) void hist_kernel(const int* __restrict__ buckets,
                                                   int* __restrict__ counts) {
    int idx = blockIdx.x * 256 + threadIdx.x;     // BH*SROW total
    int bhead = idx >> 14;
    atomicAdd(&counts[bhead * 256 + buckets[idx]], 1);
}

__global__ __launch_bounds__(256) void scan_kernel(const int* __restrict__ counts,
                                                   int* __restrict__ starts) {
    __shared__ int s[256];
    int b = blockIdx.x, t = threadIdx.x;
    int v = counts[b * 256 + t];
    s[t] = v;
    __syncthreads();
    for (int o = 1; o < 256; o <<= 1) {
        int x = (t >= o) ? s[t - o] : 0;
        __syncthreads();
        s[t] += x;
        __syncthreads();
    }
    starts[b * 256 + t] = s[t] - v;
}

// block per (bh, h); within a bucket order = t ascending (bt key structure)
__global__ __launch_bounds__(256) void rank_scatter_kernel(const int* __restrict__ buckets,
                                                           const int* __restrict__ starts,
                                                           int* __restrict__ st,
                                                           int* __restrict__ undo) {
    int bhead = blockIdx.x >> 2, h = blockIdx.x & 3;
    int tid = threadIdx.x;
    __shared__ int bb[TT];
    for (int t = tid; t < TT; t += 256)
        bb[t] = buckets[(size_t)bhead * SROW + h * TT + t];
    __syncthreads();
    for (int ch = 0; ch < 16; ++ch) {
        int t = ch * 256 + tid;
        int bk = bb[t];
        int rank = 0;
        for (int t2 = 0; t2 < t; ++t2) rank += (bb[t2] == bk);
        int pos = starts[bhead * 256 + bk] + rank;
        int j = h * TT + t;
        st  [(size_t)bhead * SROW + pos] = j;
        undo[(size_t)bhead * SROW + j]   = pos;
    }
}

// ---------------------------------------------------------------------------
// Bucketed attention: block per (bh, chunk). WMMA QK^T and P@V, LDS phase-aliased
// ---------------------------------------------------------------------------
__global__ __launch_bounds__(256) void attn_kernel(const float* __restrict__ qkproj,
                                                   const float* __restrict__ vproj,
                                                   const int* __restrict__ st,
                                                   float* __restrict__ so,
                                                   float* __restrict__ slog) {
    int gid = blockIdx.x;
    int bhead = gid >> 8, c = gid & (NCHUNK - 1);
    int b = bhead >> 3, head = bhead & 7;
    int tid = threadIdx.x, lane = tid & 31, w = tid >> 5;

    __shared__ __align__(16) unsigned short sQ[BUCKET * DH];        // 16KB (later: P)
    __shared__ __align__(16) unsigned short sK[2 * BUCKET * DH];    // 32KB (later: dots f32, Vt)
    __shared__ int bqT[BUCKET];
    __shared__ int bkT[2 * BUCKET];

    const int* strow = st + (size_t)bhead * SROW;

    // ---- gather phase: normalized K (chunk + prev chunk) and Q -------------
    if (tid < 128) {
        int j = tid;
        int srcc = (j < BUCKET) ? c : ((c + NCHUNK - 1) & (NCHUNK - 1));
        int pos = srcc * BUCKET + (j & (BUCKET - 1));
        int t = strow[pos] & (TT - 1);
        bkT[j] = t;
        const float* kr = qkproj + (size_t)(b * TT + t) * DIMD + head * DH;
        float ss = 0.0f;
        for (int d = 0; d < DH; ++d) { float v = kr[d]; ss += v * v; }
        float sc = 1.0f / fmaxf(sqrtf(ss), 1e-12f);
        for (int d = 0; d < DH; ++d) sK[j * DH + d] = f32_to_bf16(kr[d] * sc);
    } else if (tid < 192) {
        int i = tid - 128;
        int t = strow[c * BUCKET + i] & (TT - 1);
        bqT[i] = t;
        const float* qr = qkproj + (size_t)(b * TT + t) * DIMD + head * DH;
        for (int d = 0; d < DH; ++d) sQ[i * DH + d] = f32_to_bf16(qr[d]);
    }
    __syncthreads();

    // ---- dots = Q * K^T (64x128), contraction over DH ----------------------
    int mt = w & 3;            // 4 m-tiles of 16
    int ntb = (w >> 2) * 4;    // 8 n-tiles of 16, 4 per wave
    f32x8 acc[4];
#pragma unroll
    for (int j = 0; j < 4; ++j)
#pragma unroll
        for (int e = 0; e < 8; ++e) acc[j][e] = 0.0f;

    for (int k0 = 0; k0 < DH; k0 += 32) {
        bf16x16 a = load_frag_a(sQ, DH, mt * 16, k0, lane);
#pragma unroll
        for (int j = 0; j < 4; ++j) {
            bf16x16 bb = load_frag_b(sK, DH, (ntb + j) * 16, k0, lane);
            acc[j] = __builtin_amdgcn_wmma_f32_16x16x32_bf16(
                false, a, false, bb, (short)0, acc[j], false, false);
        }
    }
    __syncthreads();                       // sK reads done by all waves

    float* dots = (float*)sK;              // alias 32KB region
    int l16 = lane & 15, hi = lane >> 4;
#pragma unroll
    for (int j = 0; j < 4; ++j) {
        int colb = (ntb + j) * 16 + l16;
#pragma unroll
        for (int e = 0; e < 8; ++e)
            dots[(mt * 16 + hi * 8 + e) * 128 + colb] = acc[j][e] * SCALE_QK;
    }
    __syncthreads();

    // ---- self-mask + logsumexp softmax, P -> bf16 in sQ region -------------
    unsigned short* P = sQ;
    if (tid < BUCKET) {
        int r = tid, tq = bqT[r];
        float mx = -3.4e38f;
        for (int j2 = 0; j2 < 128; ++j2) {
            float d = dots[r * 128 + j2];
            if (bkT[j2] == tq) d = SELF_VAL;
            mx = fmaxf(mx, d);
        }
        float sum = 0.0f;
        for (int j2 = 0; j2 < 128; ++j2) {
            float d = dots[r * 128 + j2];
            if (bkT[j2] == tq) d = SELF_VAL;
            sum += expf(d - mx);
        }
        float lse = mx + logf(sum);
        for (int j2 = 0; j2 < 128; ++j2) {
            float d = dots[r * 128 + j2];
            if (bkT[j2] == tq) d = SELF_VAL;
            P[r * 128 + j2] = f32_to_bf16(expf(d - lse));
        }
        slog[(size_t)bhead * SROW + c * BUCKET + r] = lse;
    }
    __syncthreads();

    // ---- load V transposed into sK region: Vt[d][key] ----------------------
    unsigned short* Vt = sK;
    if (tid < 128) {
        int j = tid, t = bkT[j];
        const float* vr = vproj + (size_t)(b * TT + t) * DIMD + head * DH;
        for (int d = 0; d < DH; ++d) Vt[d * 128 + j] = f32_to_bf16(vr[d]);
    }
    __syncthreads();

    // ---- out = P (64x128) @ V (128x128), contraction over keys -------------
    f32x8 acc2[4];
#pragma unroll
    for (int j = 0; j < 4; ++j)
#pragma unroll
        for (int e = 0; e < 8; ++e) acc2[j][e] = 0.0f;

    for (int k0 = 0; k0 < 128; k0 += 32) {
        bf16x16 a = load_frag_a(P, 128, mt * 16, k0, lane);
#pragma unroll
        for (int j = 0; j < 4; ++j) {
            bf16x16 bb = load_frag_b(Vt, 128, (ntb + j) * 16, k0, lane);
            acc2[j] = __builtin_amdgcn_wmma_f32_16x16x32_bf16(
                false, a, false, bb, (short)0, acc2[j], false, false);
        }
    }

    float* sob = so + ((size_t)bhead * SROW + c * BUCKET) * DH;
#pragma unroll
    for (int j = 0; j < 4; ++j) {
        int colb = (ntb + j) * 16 + l16;
#pragma unroll
        for (int e = 0; e < 8; ++e)
            sob[(mt * 16 + hi * 8 + e) * DH + colb] = acc2[j][e];
    }
}

// ---------------------------------------------------------------------------
// Combine hash rounds: block per (b,t); softmax over per-round LSE logits
// ---------------------------------------------------------------------------
__global__ __launch_bounds__(256) void combine_kernel(const float* __restrict__ so,
                                                      const float* __restrict__ slog,
                                                      const int* __restrict__ undo,
                                                      unsigned short* __restrict__ attnA) {
    int bt = blockIdx.x;
    int b = bt >> 12, t = bt & (TT - 1);
    int tid = threadIdx.x;
    __shared__ int   sPos[HEADS][NHASH];
    __shared__ float sW[HEADS][NHASH];
    __shared__ float sZ[HEADS];
    if (tid < HEADS) {
        int head = tid, bhead = b * HEADS + head;
        float lg[NHASH]; int ps[NHASH];
        for (int r = 0; r < NHASH; ++r) {
            int pos = undo[(size_t)bhead * SROW + r * TT + t];
            ps[r] = pos;
            lg[r] = slog[(size_t)bhead * SROW + pos];
        }
        float m = lg[0];
        for (int r = 1; r < NHASH; ++r) m = fmaxf(m, lg[r]);
        float Z = 0.0f;
        for (int r = 0; r < NHASH; ++r) {
            float wv = expf(lg[r] - m);
            sW[head][r] = wv; sPos[head][r] = ps[r]; Z += wv;
        }
        sZ[head] = Z;
    }
    __syncthreads();
    for (int e = tid; e < DIMD; e += 256) {
        int head = e >> 7, d = e & (DH - 1);
        int bhead = b * HEADS + head;
        float a = 0.0f;
        for (int r = 0; r < NHASH; ++r)
            a += sW[head][r] * so[((size_t)bhead * SROW + sPos[head][r]) * DH + d];
        attnA[(size_t)(b * TT + t) * DIMD + e] = f32_to_bf16(a / sZ[head]);
    }
}

__global__ __launch_bounds__(256) void add_kernel(const float* __restrict__ a,
                                                  const float* __restrict__ b,
                                                  float* __restrict__ o, int n) {
    int i = blockIdx.x * 256 + threadIdx.x;
    if (i < n) o[i] = a[i] + b[i];
}

// ---------------------------------------------------------------------------
extern "C" void kernel_launch(void* const* d_in, const int* in_sizes, int n_in,
                              void* d_out, int out_size, void* d_ws, size_t ws_size,
                              hipStream_t stream) {
    (void)in_sizes; (void)n_in; (void)out_size; (void)ws_size;
    const float* x    = (const float*)d_in[0];
    const float* ln1g = (const float*)d_in[1];
    const float* ln1b = (const float*)d_in[2];
    const float* Wqk  = (const float*)d_in[3];
    const float* Wv   = (const float*)d_in[4];
    const float* Wo   = (const float*)d_in[5];
    const float* bo   = (const float*)d_in[6];
    const float* ln2g = (const float*)d_in[7];
    const float* ln2b = (const float*)d_in[8];
    const float* W1   = (const float*)d_in[9];
    const float* b1   = (const float*)d_in[10];
    const float* W2   = (const float*)d_in[11];
    const float* b2   = (const float*)d_in[12];
    const float* rot  = (const float*)d_in[13];
    float* outp = (float*)d_out;

    char* wsb = (char*)d_ws; size_t off = 0;
    auto alloc = [&](size_t bytes) -> void* {
        off = (off + 255) & ~(size_t)255;
        void* p = wsb + off; off += bytes; return p;
    };

    unsigned short *wqkT[2], *wvT[2], *woT[2], *w1T[2], *w2T[2];
    for (int l = 0; l < 2; ++l) {
        wqkT[l] = (unsigned short*)alloc((size_t)DIMD * DIMD * 2);
        wvT [l] = (unsigned short*)alloc((size_t)DIMD * DIMD * 2);
        woT [l] = (unsigned short*)alloc((size_t)DIMD * DIMD * 2);
        w1T [l] = (unsigned short*)alloc((size_t)DIMD * 4 * DIMD * 2);
        w2T [l] = (unsigned short*)alloc((size_t)DIMD * 4 * DIMD * 2);
    }
    unsigned short* xnbf  = (unsigned short*)alloc((size_t)MTOK * DIMD * 2);
    float* qkproj = (float*)alloc((size_t)MTOK * DIMD * 4);
    float* vproj  = (float*)alloc((size_t)MTOK * DIMD * 4);
    int*   buckets= (int*)alloc((size_t)BH * SROW * 4);
    int*   counts = (int*)alloc((size_t)BH * 256 * 4);
    int*   starts = (int*)alloc((size_t)BH * 256 * 4);
    int*   stb    = (int*)alloc((size_t)BH * SROW * 4);
    int*   undo   = (int*)alloc((size_t)BH * SROW * 4);
    float* so     = (float*)alloc((size_t)BH * SROW * DH * 4);
    float* slog   = (float*)alloc((size_t)BH * SROW * 4);
    unsigned short* attnA = (unsigned short*)alloc((size_t)MTOK * DIMD * 2);
    unsigned short* hbf   = (unsigned short*)alloc((size_t)MTOK * DIMD * 2);
    unsigned short* ff1   = (unsigned short*)alloc((size_t)MTOK * 4 * DIMD * 2);
    float* y1[2], *y2[2];
    for (int l = 0; l < 2; ++l) {
        y1[l] = (float*)alloc((size_t)MTOK * DIMD * 4);
        y2[l] = (float*)alloc((size_t)MTOK * DIMD * 4);
    }

    // weight conversion (f32 [K][N] -> bf16 [N][K])
    for (int l = 0; l < 2; ++l) {
        int n1 = DIMD * DIMD, n2 = DIMD * 4 * DIMD;
        convt_kernel<<<(n1 + 255) / 256, 256, 0, stream>>>(Wqk + (size_t)l * n1, wqkT[l], DIMD, DIMD);
        convt_kernel<<<(n1 + 255) / 256, 256, 0, stream>>>(Wv  + (size_t)l * n1, wvT [l], DIMD, DIMD);
        convt_kernel<<<(n1 + 255) / 256, 256, 0, stream>>>(Wo  + (size_t)l * n1, woT [l], DIMD, DIMD);
        convt_kernel<<<(n2 + 255) / 256, 256, 0, stream>>>(W1  + (size_t)l * n2, w1T [l], DIMD, 4 * DIMD);
        convt_kernel<<<(n2 + 255) / 256, 256, 0, stream>>>(W2  + (size_t)l * n2, w2T [l], 4 * DIMD, DIMD);
    }

    const float* x1 = x;
    const float* x2 = x;
    for (int l = 0; l < 2; ++l) {
        // attention branch: y1 = x1 + Attn(LN(x2))
        ln_kernel<<<MTOK, 256, 0, stream>>>(x2, ln1g + l * DIMD, ln1b + l * DIMD, xnbf);
        gemm_bf16_kernel<<<dim3(DIMD / 128, MTOK / 128), 256, 0, stream>>>(
            xnbf, wqkT[l], nullptr, nullptr, qkproj, nullptr, MTOK, DIMD, DIMD, 0);
        gemm_bf16_kernel<<<dim3(DIMD / 128, MTOK / 128), 256, 0, stream>>>(
            xnbf, wvT[l], nullptr, nullptr, vproj, nullptr, MTOK, DIMD, DIMD, 0);
        hash_kernel<<<BH * TT, 128, 0, stream>>>(qkproj, rot + (size_t)l * DH * NHASH * 32, buckets);
        (void)hipMemsetAsync(counts, 0, (size_t)BH * 256 * 4, stream);
        hist_kernel<<<(BH * SROW) / 256, 256, 0, stream>>>(buckets, counts);
        scan_kernel<<<BH, 256, 0, stream>>>(counts, starts);
        rank_scatter_kernel<<<BH * NHASH, 256, 0, stream>>>(buckets, starts, stb, undo);
        attn_kernel<<<BH * NCHUNK, 256, 0, stream>>>(qkproj, vproj, stb, so, slog);
        combine_kernel<<<MTOK, 256, 0, stream>>>(so, slog, undo, attnA);
        gemm_bf16_kernel<<<dim3(DIMD / 128, MTOK / 128), 256, 0, stream>>>(
            attnA, woT[l], bo + l * DIMD, x1, y1[l], nullptr, MTOK, DIMD, DIMD, 0);
        // FFN branch: y2 = x2 + W2(gelu(W1(LN(y1))+b1))+b2
        ln_kernel<<<MTOK, 256, 0, stream>>>(y1[l], ln2g + l * DIMD, ln2b + l * DIMD, hbf);
        gemm_bf16_kernel<<<dim3(4 * DIMD / 128, MTOK / 128), 256, 0, stream>>>(
            hbf, w1T[l], b1 + l * 4 * DIMD, nullptr, nullptr, ff1, MTOK, 4 * DIMD, DIMD, 1);
        gemm_bf16_kernel<<<dim3(DIMD / 128, MTOK / 128), 256, 0, stream>>>(
            ff1, w2T[l], b2 + l * DIMD, x2, y2[l], nullptr, MTOK, DIMD, 4 * DIMD, 0);
        x1 = y1[l];
        x2 = y2[l];
    }
    add_kernel<<<(MTOK * DIMD + 255) / 256, 256, 0, stream>>>(x1, x2, outp, MTOK * DIMD);
}
